// DGCNN_GAT_54700703481981
// MI455X (gfx1250) — matrix operand order repaired and checked by reading, not compiled, not run on hardware
//
#include <hip/hip_runtime.h>
#include <hip/hip_bf16.h>

// ---------------- problem constants (from reference) ----------------
#define NB   256      // batches
#define NE   64       // points per batch
#define NC   32       // channels
#define NPTS (NB*NE)  // 16384 total points
#define KNN  4
#define NHEADS 4
#define NHID   32
#define HOUT  (NHEADS*NHID)   // 128
#define NEG_SLOPE 0.2f

// ---------------- WMMA types (CDNA5, wave32) ----------------
typedef __bf16 bf16_t;
typedef bf16_t v16bf __attribute__((ext_vector_type(16)));
typedef float  v8f   __attribute__((ext_vector_type(8)));

// =====================================================================
// Kernel 1: f32 -> bf16 (RNE) conversion of the point cloud
// =====================================================================
__global__ __launch_bounds__(256) void dg_convert_bf16(
    const float* __restrict__ x, unsigned short* __restrict__ xb, int n) {
  int i = blockIdx.x * 256 + threadIdx.x;
  if (i < n) {
    unsigned u = __float_as_uint(x[i]);
    unsigned r = u + 0x7FFFu + ((u >> 16) & 1u);   // round-to-nearest-even
    xb[i] = (unsigned short)(r >> 16);
  }
}

// =====================================================================
// Kernel 2: squared norms (kept in f32 for distance accuracy)
// =====================================================================
__global__ __launch_bounds__(256) void dg_sqnorm(
    const float* __restrict__ x, float* __restrict__ sqn) {
  int n = blockIdx.x * 256 + threadIdx.x;   // exactly NPTS threads
  const float* r = x + n * NC;
  float s = 0.f;
#pragma unroll
  for (int c = 0; c < NC; ++c) { float v = r[c]; s += v * v; }
  sqn[n] = s;
}

// =====================================================================
// Kernel 3: tiled WMMA KNN.  128 rows per block, 4 waves.
//  - wave w owns row-bands {2w, 2w+1}; A fragments loaded once
//  - column loop of 32: 2 B fragments, 4 x v_wmma_f32_16x16x32_bf16
//  - distances staged in LDS (stride 33), each thread keeps top-4
// =====================================================================
__global__ __launch_bounds__(128) void dg_knn_wmma(
    const unsigned short* __restrict__ xb_us,
    const float* __restrict__ sqn,
    int* __restrict__ idx_out) {
  const bf16_t* xb = reinterpret_cast<const bf16_t*>(xb_us);

  __shared__ float dist[128][33];   // 16.5 KB, 33-stride => conflict-free row scan
  __shared__ float rowSq[128];

  const int t     = threadIdx.x;
  const int w     = t >> 5;         // wave id 0..3 (wave32)
  const int lane  = t & 31;
  const int lo16  = lane & 15;
  const bool hi   = lane >= 16;
  const int rowBlock = blockIdx.x * 128;
  const int myRow    = rowBlock + t;

  rowSq[t] = sqn[myRow];

  // ---- A fragments: 16x32 bf16 in contiguous k order (k = 16*(lane/16) + s), same as B ----
  // lane<16 : row M=lane,   K = 0..15
  // lane>=16: row M=lane-16,K = 16..31
  v16bf afrag[2];
#pragma unroll
  for (int rb = 0; rb < 2; ++rb) {
    int row = rowBlock + (2 * w + rb) * 16 + lo16;
    const bf16_t* rp = xb + row * NC + (hi ? 16 : 0);
#pragma unroll
    for (int i = 0; i < 8; ++i) afrag[rb][i] = rp[i];
#pragma unroll
    for (int i = 0; i < 8; ++i) afrag[rb][8 + i] = rp[8 + i];
  }

  // running top-4 (ascending distance), matches top_k(-d) ordering
  float d0 = 3.4e38f, d1 = 3.4e38f, d2 = 3.4e38f, d3 = 3.4e38f;
  int   i0 = 0, i1 = 0, i2 = 0, i3 = 0;

  __syncthreads();

  for (int colBase = 0; colBase < NPTS; colBase += 32) {
    // ---- compute phase: 4 WMMA tiles per wave ----
#pragma unroll
    for (int cb = 0; cb < 2; ++cb) {
      int col = colBase + cb * 16 + lo16;
      // B 32x16 bf16: lane<16 holds col N=lane, K=0..15; lane>=16 K=16..31
      const v16bf bfrag =
          *reinterpret_cast<const v16bf*>(xb + col * NC + (hi ? 16 : 0));
      float sqc = sqn[col];
      int lc = cb * 16 + lo16;
#pragma unroll
      for (int rb = 0; rb < 2; ++rb) {
        v8f acc = {0.f, 0.f, 0.f, 0.f, 0.f, 0.f, 0.f, 0.f};
        acc = __builtin_amdgcn_wmma_f32_16x16x32_bf16(
            false, afrag[rb], false, bfrag, (short)0, acc, false, false);
        int rbase = (2 * w + rb) * 16 + (hi ? 8 : 0);
#pragma unroll
        for (int i = 0; i < 8; ++i) {
          dist[rbase + i][lc] = __builtin_fmaf(-2.0f, acc[i],
                                               rowSq[rbase + i] + sqc);
        }
      }
    }
    __syncthreads();

    // ---- selection phase: thread t scans its row's 32 columns ----
#pragma unroll 8
    for (int c = 0; c < 32; ++c) {
      int g = colBase + c;
      float d = dist[t][c];
      if (g != myRow && d < d3) {           // diagonal masked like d[r,r]=inf
        if (d < d2) {
          d3 = d2; i3 = i2;
          if (d < d1) {
            d2 = d1; i2 = i1;
            if (d < d0) { d1 = d0; i1 = i0; d0 = d; i0 = g; }
            else        { d1 = d;  i1 = g; }
          } else        { d2 = d;  i2 = g; }
        } else          { d3 = d;  i3 = g; }
      }
    }
    __syncthreads();
  }

  idx_out[myRow * KNN + 0] = i0;
  idx_out[myRow * KNN + 1] = i1;
  idx_out[myRow * KNN + 2] = i2;
  idx_out[myRow * KNN + 3] = i3;
}

// =====================================================================
// Kernel 4: h = x @ gat_w   (16384x32 @ 32x128), f32
// =====================================================================
__global__ __launch_bounds__(256) void dg_hproj(
    const float* __restrict__ x, const float* __restrict__ W,
    float* __restrict__ h) {
  int id = blockIdx.x * 256 + threadIdx.x;   // NPTS*HOUT threads exactly
  int n = id >> 7, j = id & (HOUT - 1);
  const float* r = x + n * NC;
  float s = 0.f;
#pragma unroll
  for (int c = 0; c < NC; ++c) s = __builtin_fmaf(r[c], W[c * HOUT + j], s);
  h[id] = s;
}

// =====================================================================
// Kernel 5: per-(node,head) attention logits asrc/adst
// =====================================================================
__global__ __launch_bounds__(256) void dg_attn_logits(
    const float* __restrict__ h, const float* __restrict__ a_src,
    const float* __restrict__ a_dst, float* __restrict__ asrc,
    float* __restrict__ adst) {
  int id = blockIdx.x * 256 + threadIdx.x;   // NPTS*NHEADS threads
  int n = id >> 2, hd = id & (NHEADS - 1);
  const float* hp = h + n * HOUT + hd * NHID;
  const float* as = a_src + hd * NHID;
  const float* ad = a_dst + hd * NHID;
  float s = 0.f, d = 0.f;
#pragma unroll
  for (int c = 0; c < NHID; ++c) {
    float v = hp[c];
    s = __builtin_fmaf(v, as[c], s);
    d = __builtin_fmaf(v, ad[c], d);
  }
  asrc[id] = s;
  adst[id] = d;
}

// =====================================================================
// Kernel 6: GAT softmax-aggregate over {4 knn + self}, +bias, ELU
// =====================================================================
__global__ __launch_bounds__(256) void dg_gat_aggregate(
    const float* __restrict__ h, const int* __restrict__ idx,
    const float* __restrict__ asrc, const float* __restrict__ adst,
    const float* __restrict__ gat_b, float* __restrict__ agg) {
  int id = blockIdx.x * 256 + threadIdx.x;   // NPTS*HOUT threads
  int n = id >> 7, j = id & (HOUT - 1);
  int hd = j >> 5;

  int nb[KNN + 1];
#pragma unroll
  for (int k = 0; k < KNN; ++k) nb[k] = idx[n * KNN + k];
  nb[KNN] = n;                               // self appended last (reference order)

  float ad = adst[n * NHEADS + hd];
  float e[KNN + 1];
  float m = -3.4e38f;
#pragma unroll
  for (int k = 0; k <= KNN; ++k) {
    float v = ad + asrc[nb[k] * NHEADS + hd];
    v = v > 0.f ? v : NEG_SLOPE * v;         // leaky_relu
    e[k] = v;
    m = v > m ? v : m;
  }
  float ssum = 0.f, acc = 0.f;
#pragma unroll
  for (int k = 0; k <= KNN; ++k) {
    float wgt = __expf(e[k] - m);
    ssum += wgt;
    acc = __builtin_fmaf(wgt, h[nb[k] * HOUT + j], acc);
  }
  float o = acc / ssum + gat_b[j];
  o = o > 0.f ? o : (__expf(o) - 1.0f);      // elu
  agg[id] = o;
}

// =====================================================================
// Kernel 7: mean over E=64 within each batch, then @ fc_w + fc_b
// one block per batch, 128 threads (= HOUT)
// =====================================================================
__global__ __launch_bounds__(128) void dg_mean_fc(
    const float* __restrict__ agg, const float* __restrict__ fc_w,
    const float* __restrict__ fc_b, float* __restrict__ out) {
  __shared__ float m[HOUT];
  int b = blockIdx.x, t = threadIdx.x;
  float s = 0.f;
  const float* base = agg + (size_t)b * NE * HOUT + t;
#pragma unroll 8
  for (int e = 0; e < NE; ++e) s += base[e * HOUT];
  m[t] = s * (1.0f / NE);
  __syncthreads();
  if (t < 2) {
    float acc = fc_b[t];
#pragma unroll 8
    for (int c = 0; c < HOUT; ++c) acc = __builtin_fmaf(m[c], fc_w[c * 2 + t], acc);
    out[b * 2 + t] = acc;
  }
}

// =====================================================================
// launcher
// =====================================================================
extern "C" void kernel_launch(void* const* d_in, const int* in_sizes, int n_in,
                              void* d_out, int out_size, void* d_ws, size_t ws_size,
                              hipStream_t stream) {
  const float* x     = (const float*)d_in[0];
  // d_in[1..4] = edge-MLP params: dead code (never reaches the output)
  const float* gat_w = (const float*)d_in[5];
  const float* a_src = (const float*)d_in[6];
  const float* a_dst = (const float*)d_in[7];
  const float* gat_b = (const float*)d_in[8];
  const float* fc_w  = (const float*)d_in[9];
  const float* fc_b  = (const float*)d_in[10];
  float* out = (float*)d_out;

  char* ws = (char*)d_ws;
  size_t off = 0;
  auto alloc = [&](size_t bytes) -> void* {
    void* p = ws + off;
    off = (off + bytes + 255) & ~(size_t)255;
    return p;
  };
  unsigned short* xb  = (unsigned short*)alloc((size_t)NPTS * NC * 2);  // 1 MB bf16
  float* sqn          = (float*)alloc((size_t)NPTS * 4);
  int*   idx          = (int*)  alloc((size_t)NPTS * KNN * 4);
  float* hbuf         = (float*)alloc((size_t)NPTS * HOUT * 4);         // 8 MB
  float* asrc         = (float*)alloc((size_t)NPTS * NHEADS * 4);
  float* adst         = (float*)alloc((size_t)NPTS * NHEADS * 4);
  float* agg          = (float*)alloc((size_t)NPTS * HOUT * 4);         // 8 MB

  dg_convert_bf16<<<(NPTS * NC + 255) / 256, 256, 0, stream>>>(x, xb, NPTS * NC);
  dg_sqnorm<<<NPTS / 256, 256, 0, stream>>>(x, sqn);
  dg_knn_wmma<<<NPTS / 128, 128, 0, stream>>>(xb, sqn, idx);
  dg_hproj<<<(NPTS * HOUT) / 256, 256, 0, stream>>>(x, gat_w, hbuf);
  dg_attn_logits<<<(NPTS * NHEADS) / 256, 256, 0, stream>>>(hbuf, a_src, a_dst, asrc, adst);
  dg_gat_aggregate<<<(NPTS * HOUT) / 256, 256, 0, stream>>>(hbuf, idx, asrc, adst, gat_b, agg);
  dg_mean_fc<<<NB, HOUT, 0, stream>>>(agg, fc_w, fc_b, out);
}
